// Attention_88029649699625
// MI455X (gfx1250) — matrix operand order repaired and checked by reading, not compiled, not run on hardware
//
#include <hip/hip_runtime.h>
#include <hip/hip_bf16.h>
#include <cstdint>
#include <cstddef>

// Sizes from the reference
#define NHID   1024
#define HEADS  16
#define SEQ    2048
#define BATCH  4
#define DIMH   64            // NHID / HEADS
#define MROWS  (SEQ * BATCH) // 8192 GEMM rows

typedef _Float16 half_t;
typedef __attribute__((ext_vector_type(16))) _Float16 v16h;
typedef __attribute__((ext_vector_type(8)))  _Float16 v8h;
typedef __attribute__((ext_vector_type(8)))  float    v8f;
typedef __attribute__((ext_vector_type(4)))  float    v4f;
typedef __attribute__((ext_vector_type(4)))  int      v4i;

// ---------------------------------------------------------------------------
// CDNA5 async global->LDS copy (ASYNCcnt-tracked), guarded by __has_builtin so
// the file still compiles (with the synchronous fallback) on toolchains that
// do not expose the builtin.  Probe round 2: builtin exists; param 1 is
// 'v4i __device__ *' (addrspace(1) int4 pointer).
// ---------------------------------------------------------------------------
#if defined(__has_builtin)
#  if __has_builtin(__builtin_amdgcn_global_load_async_to_lds_b128)
#    define HAVE_ASYNC_LDS 1
#  endif
#endif
#ifndef HAVE_ASYNC_LDS
#  define HAVE_ASYNC_LDS 0
#endif

#if HAVE_ASYNC_LDS
typedef v4i __attribute__((address_space(1)))* gvi_t;   // global int4*
typedef v4i __attribute__((address_space(3)))* lvi_t;   // LDS int4*
__device__ __forceinline__ void async_b128(void* lds, const void* g) {
    __builtin_amdgcn_global_load_async_to_lds_b128((gvi_t)g, (lvi_t)lds,
                                                   /*offset=*/0, /*cpol=*/0);
}
__device__ __forceinline__ void wait_async0() {
#  if __has_builtin(__builtin_amdgcn_s_wait_asynccnt)
    __builtin_amdgcn_s_wait_asynccnt(0);
#  else
    asm volatile("s_wait_asynccnt 0x0" ::: "memory");
#  endif
}
#else
__device__ __forceinline__ void wait_async0() {}
#endif

__device__ __forceinline__ v8f wmma32(v16h a, v16h b, v8f c) {
    // D = A(16x32 f16) * B(32x16 f16) + C(16x16 f32)
    return __builtin_amdgcn_wmma_f32_16x16x32_f16(
        /*neg_a=*/false, a, /*neg_b=*/false, b,
        /*c_mod=*/(short)0, c, /*reuse_a=*/false, /*reuse_b=*/false);
}

// A-operand: lane holds row (lane&15); element j<8 -> K = hlf*8 + j,
// element j>=8 -> K = 16 + hlf*8 + (j-8).   (ISA 7.12.2 16-bit A 16x32)
__device__ __forceinline__ v16h fragA(const half_t* p, int hlf) {
    v8h lo = *(const v8h*)(p + hlf * 8);
    v8h hi = *(const v8h*)(p + 16 + hlf * 8);
    v16h r;
#pragma unroll
    for (int i = 0; i < 8; ++i) { r[i] = lo[i]; r[i + 8] = hi[i]; }
    return r;
}

// B-operand: lane holds column (lane&15); element j<8 -> K = hlf*8 + j,
// element j>=8 -> K = 16 + hlf*8 + (j-8)   (same order as A)
__device__ __forceinline__ v16h fragB(const half_t* p, int hlf) {
    v8h lo = *(const v8h*)(p + hlf * 8);
    v8h hi = *(const v8h*)(p + 16 + hlf * 8);
    v16h r;
#pragma unroll
    for (int i = 0; i < 8; ++i) { r[i] = lo[i]; r[i + 8] = hi[i]; }
    return r;
}

__device__ __forceinline__ float sigmoidf_(float x) { return 1.0f / (1.0f + __expf(-x)); }

// ---------------------------------------------------------------------------
// f32 -> f16 weight conversion
// ---------------------------------------------------------------------------
__global__ void cvt_f32_f16(const float* __restrict__ in, half_t* __restrict__ out, int n) {
    int i = blockIdx.x * blockDim.x + threadIdx.x;
    int stride = gridDim.x * blockDim.x;
    for (; i < n; i += stride) out[i] = (half_t)in[i];
}

// ---------------------------------------------------------------------------
// Gate chain: cf[j] = sigmoid(vs_p) . vq_w[j,:] + vq_b[j]   (j in [0,2048))
// ---------------------------------------------------------------------------
__global__ void gate_cf_kernel(const float* __restrict__ vs_p,
                               const float* __restrict__ vq_w,
                               const float* __restrict__ vq_b,
                               float* __restrict__ cf) {
    __shared__ float red[256];
    const int j = blockIdx.x;
    float s = 0.0f;
    for (int i = threadIdx.x; i < NHID; i += 256)
        s += sigmoidf_(vs_p[i]) * vq_w[(size_t)j * NHID + i];
    red[threadIdx.x] = s;
    __syncthreads();
    for (int st = 128; st > 0; st >>= 1) {
        if (threadIdx.x < st) red[threadIdx.x] += red[threadIdx.x + st];
        __syncthreads();
    }
    if (threadIdx.x == 0) cf[j] = red[0] + vq_b[j];
}

// qs = sigmoid(qs_p), ks = sigmoid(ks_p), vs = sigmoid(f)*tanh(c)
__global__ void gate_scales_kernel(const float* __restrict__ cf,
                                   const float* __restrict__ qs_p,
                                   const float* __restrict__ ks_p,
                                   float* __restrict__ qs, float* __restrict__ ks,
                                   float* __restrict__ vs) {
    int n = blockIdx.x * blockDim.x + threadIdx.x;
    if (n < NHID) {
        qs[n] = sigmoidf_(qs_p[n]);
        ks[n] = sigmoidf_(ks_p[n]);
        float c = cf[n], f = cf[NHID + n];
        vs[n] = sigmoidf_(f) * tanhf(c);
    }
}

// ---------------------------------------------------------------------------
// WMMA GEMM: out = epi( A(MxK) * W(NtotxK)^T )
// 256 threads = 8 waves; block tile 128x128; wave tile 32x64; K-step 32.
// AF16=false: A is f32 (converted while staging).  FINAL: gelu+residual epi.
// f16 tiles are staged with GLOBAL_LOAD_ASYNC_TO_LDS when available.
// ---------------------------------------------------------------------------
template <bool AF16, bool FINAL>
__global__ __launch_bounds__(256)
void gemm_wmma_kernel(const void* __restrict__ A, const half_t* __restrict__ W,
                      const float* __restrict__ bias, const float* __restrict__ scale,
                      const float* __restrict__ mixf, half_t* __restrict__ out16,
                      float* __restrict__ out32, int K, int ostride) {
    const int tid = threadIdx.x;
    const int wave = tid >> 5, lane = tid & 31;
    const int lane16 = lane & 15, hlf = lane >> 4;
    const int wm = (wave & 3) * 32;   // wave M offset in tile
    const int wn = (wave >> 2) * 64;  // wave N offset in tile
    const int M0 = blockIdx.y * 128, N0 = blockIdx.x * 128;

    __shared__ __align__(16) half_t Ash[128 * 32];
    __shared__ __align__(16) half_t Bsh[128 * 32];

    v8f acc[2][4];
#pragma unroll
    for (int mt = 0; mt < 2; ++mt)
#pragma unroll
        for (int nt = 0; nt < 4; ++nt) acc[mt][nt] = {};

    const int arow = tid & 127;          // staging row in tile
    const int acol = (tid >> 7) * 16;    // staging col: 16 halves per thread

    for (int k0 = 0; k0 < K; k0 += 32) {
        __syncthreads();
        // ---- stage A tile (128x32 f16) ----
        if constexpr (AF16) {
            const half_t* Arow = (const half_t*)A + (size_t)(M0 + arow) * K + k0 + acol;
#if HAVE_ASYNC_LDS
            async_b128(Ash + arow * 32 + acol, Arow);
            async_b128(Ash + arow * 32 + acol + 8, Arow + 8);
#else
            __builtin_prefetch((const char*)Arow + 64, 0, 1);
            *(v8h*)(Ash + arow * 32 + acol)     = *(const v8h*)Arow;
            *(v8h*)(Ash + arow * 32 + acol + 8) = *(const v8h*)(Arow + 8);
#endif
        } else {
            const float* Arow = (const float*)A + (size_t)(M0 + arow) * K + k0 + acol;
            __builtin_prefetch((const char*)Arow + 128, 0, 1);
            __align__(16) half_t tmp[16];
#pragma unroll
            for (int i = 0; i < 4; ++i) {
                v4f f = *(const v4f*)(Arow + i * 4);
#pragma unroll
                for (int j = 0; j < 4; ++j) tmp[i * 4 + j] = (half_t)f[j];
            }
            *(v8h*)(Ash + arow * 32 + acol)     = *(const v8h*)tmp;
            *(v8h*)(Ash + arow * 32 + acol + 8) = *(const v8h*)(tmp + 8);
        }
        // ---- stage W tile (128 rows of W, 32 k) ----
        {
            const half_t* Brow = W + (size_t)(N0 + arow) * K + k0 + acol;
#if HAVE_ASYNC_LDS
            async_b128(Bsh + arow * 32 + acol, Brow);
            async_b128(Bsh + arow * 32 + acol + 8, Brow + 8);
#else
            __builtin_prefetch((const char*)Brow + 64, 0, 1);
            *(v8h*)(Bsh + arow * 32 + acol)     = *(const v8h*)Brow;
            *(v8h*)(Bsh + arow * 32 + acol + 8) = *(const v8h*)(Brow + 8);
#endif
        }
        wait_async0();
        __syncthreads();
        // ---- compute ----
        v16h af[2], bf[4];
#pragma unroll
        for (int mt = 0; mt < 2; ++mt)
            af[mt] = fragA(Ash + (wm + mt * 16 + lane16) * 32, hlf);
#pragma unroll
        for (int nt = 0; nt < 4; ++nt)
            bf[nt] = fragB(Bsh + (wn + nt * 16 + lane16) * 32, hlf);
#pragma unroll
        for (int mt = 0; mt < 2; ++mt)
#pragma unroll
            for (int nt = 0; nt < 4; ++nt)
                acc[mt][nt] = wmma32(af[mt], bf[nt], acc[mt][nt]);
    }

    // ---- epilogue ----
#pragma unroll
    for (int mt = 0; mt < 2; ++mt)
#pragma unroll
        for (int nt = 0; nt < 4; ++nt)
#pragma unroll
            for (int r = 0; r < 8; ++r) {
                const int m = M0 + wm + mt * 16 + r + hlf * 8;
                const int n = N0 + wn + nt * 16 + lane16;
                const float a = acc[mt][nt][r];
                if constexpr (FINAL) {
                    float rr = a + bias[n];
                    float g  = rr * sigmoidf_(1.702f * rr);     // gelu_bert
                    float gt = sigmoidf_(scale[n]);             // sigmoid(r_gate)
                    out32[(size_t)m * NHID + n] = gt * mixf[(size_t)m * NHID + n] + g;
                } else {
                    float y = scale[n] * (a + bias[n]);
                    out16[(size_t)m * ostride + n] = (half_t)y;
                }
            }
}

// ---------------------------------------------------------------------------
// Flash attention: grid (SEQ/64, BATCH*HEADS), 128 threads = 4 waves,
// wave owns 16 query rows; K/V tiles of 128 in LDS; online softmax in regs.
// K tile is async-staged; V tile is transposed through VGPRs.
// ---------------------------------------------------------------------------
__global__ __launch_bounds__(128)
void attn_kernel(const half_t* __restrict__ q16,  // = rcat + 1024, row stride 2048
                 const half_t* __restrict__ k16,  // (8192,1024) f16
                 const half_t* __restrict__ v16,  // (8192,1024) f16
                 half_t* __restrict__ mix16,      // = rcat, row stride 2048
                 float* __restrict__ mixf) {      // (8192,1024) f32
    const int tid = threadIdx.x;
    const int wave = tid >> 5, lane = tid & 31;
    const int lane16 = lane & 15, hlf = lane >> 4;
    const int bh = blockIdx.y;
    const int b = bh >> 4, h = bh & 15;
    const int qbase = blockIdx.x * 64 + wave * 16;

    __shared__ __align__(16) half_t KP[128 * 64];   // K tile; reused as P (4 waves * 16x128)
    __shared__ __align__(16) half_t Vt[64 * 136];   // V^T tile, padded stride 136

    // load Q operand frags once (d = 0..63 -> two K=32 frags)
    const half_t* qrow = q16 + ((size_t)(qbase + lane16) * BATCH + b) * (2 * NHID) + h * DIMH;
    v16h qa0 = fragA(qrow, hlf);
    v16h qa1 = fragA(qrow + 32, hlf);

    float mrow[8], lrow[8];
    v8f oacc[4];
#pragma unroll
    for (int r = 0; r < 8; ++r) { mrow[r] = -1e30f; lrow[r] = 0.0f; }
#pragma unroll
    for (int dt = 0; dt < 4; ++dt) oacc[dt] = {};

    half_t* Pw = KP + wave * 16 * 128;

    for (int kt = 0; kt < SEQ; kt += 128) {
        __syncthreads();
        // ---- stage K tile: one 64-half row per thread ----
        {
            const half_t* krow = k16 + ((size_t)(kt + tid) * BATCH + b) * NHID + h * DIMH;
#if HAVE_ASYNC_LDS
#pragma unroll
            for (int i = 0; i < 8; ++i)
                async_b128(KP + tid * 64 + i * 8, krow + i * 8);
#else
#pragma unroll
            for (int i = 0; i < 8; ++i)
                *(v8h*)(KP + tid * 64 + i * 8) = *(const v8h*)(krow + i * 8);
#endif
        }
        // ---- stage V tile transposed: Vt[d][kk] ----
        {
            const half_t* vrow = v16 + ((size_t)(kt + tid) * BATCH + b) * NHID + h * DIMH;
            half_t vloc[64];
#pragma unroll
            for (int i = 0; i < 8; ++i)
                *(v8h*)(vloc + i * 8) = *(const v8h*)(vrow + i * 8);
#pragma unroll
            for (int d = 0; d < 64; ++d) Vt[d * 136 + tid] = vloc[d];
        }
        wait_async0();
        __syncthreads();

        // ---- S = Q K^T (16 x 128), 8 N-tiles x 2 K-steps ----
        v8f sc[8];
#pragma unroll
        for (int nt = 0; nt < 8; ++nt) {
            sc[nt] = {};
            const half_t* kb = KP + (nt * 16 + lane16) * 64;
            sc[nt] = wmma32(qa0, fragB(kb, hlf), sc[nt]);
            sc[nt] = wmma32(qa1, fragB(kb + 32, hlf), sc[nt]);
        }

        __syncthreads();   // everyone done reading K tile -> KP reusable as P

        // ---- online softmax (registers + wave32 shuffles) + write P ----
#pragma unroll
        for (int r = 0; r < 8; ++r) {
            float pm = -1e30f;
#pragma unroll
            for (int nt = 0; nt < 8; ++nt) pm = fmaxf(pm, sc[nt][r]);
            pm *= 0.125f;  // 1/sqrt(64)
#pragma unroll
            for (int off = 1; off < 16; off <<= 1)
                pm = fmaxf(pm, __shfl_xor(pm, off, 32));
            float mnew  = fmaxf(mrow[r], pm);
            float alpha = exp2f((mrow[r] - mnew) * 1.44269504f);
            mrow[r] = mnew;
            lrow[r] *= alpha;
#pragma unroll
            for (int dt = 0; dt < 4; ++dt) oacc[dt][r] *= alpha;
            float rs = 0.0f;
            const int prow = r + hlf * 8;
#pragma unroll
            for (int nt = 0; nt < 8; ++nt) {
                float p = exp2f((sc[nt][r] * 0.125f - mnew) * 1.44269504f);
                rs += p;
                Pw[prow * 128 + nt * 16 + lane16] = (half_t)p;
            }
#pragma unroll
            for (int off = 1; off < 16; off <<= 1)
                rs += __shfl_xor(rs, off, 32);
            lrow[r] += rs;
        }
        __syncthreads();   // P visible

        // ---- O += P (16x128) * V (128x64): 4 d-tiles x 4 K-steps ----
        v16h pa[4];
#pragma unroll
        for (int ks = 0; ks < 4; ++ks)
            pa[ks] = fragA(Pw + lane16 * 128 + ks * 32, hlf);
#pragma unroll
        for (int dt = 0; dt < 4; ++dt)
#pragma unroll
            for (int ks = 0; ks < 4; ++ks)
                oacc[dt] = wmma32(pa[ks], fragB(Vt + (dt * 16 + lane16) * 136 + ks * 32, hlf),
                                  oacc[dt]);
    }

    // ---- normalize and write mix (f32 residual copy + f16 into rcat) ----
#pragma unroll
    for (int r = 0; r < 8; ++r) {
        const float inv = (lrow[r] > 0.0f) ? (1.0f / lrow[r]) : 0.0f;
        const int s = qbase + r + hlf * 8;
        const size_t rowm = (size_t)s * BATCH + b;
#pragma unroll
        for (int dt = 0; dt < 4; ++dt) {
            const int d = dt * 16 + lane16;
            const float val = oacc[dt][r] * inv;
            mixf[rowm * NHID + h * DIMH + d]        = val;
            mix16[rowm * (2 * NHID) + h * DIMH + d] = (half_t)val;
        }
    }
}

// ---------------------------------------------------------------------------
// Host launch.  Workspace requirement: ~106 MB (see offsets below).
// ---------------------------------------------------------------------------
extern "C" void kernel_launch(void* const* d_in, const int* in_sizes, int n_in,
                              void* d_out, int out_size, void* d_ws, size_t ws_size,
                              hipStream_t stream) {
    (void)in_sizes; (void)n_in; (void)out_size; (void)ws_size;
    const float* query = (const float*)d_in[0];
    const float* key   = (const float*)d_in[1];
    const float* value = (const float*)d_in[2];
    const float* qs_p  = (const float*)d_in[3];
    const float* ks_p  = (const float*)d_in[4];
    const float* vs_p  = (const float*)d_in[5];
    const float* vq_w  = (const float*)d_in[6];
    const float* vq_b  = (const float*)d_in[7];
    const float* q_w   = (const float*)d_in[8];
    const float* q_b   = (const float*)d_in[9];
    const float* k_w   = (const float*)d_in[10];
    const float* k_b   = (const float*)d_in[11];
    const float* v_w   = (const float*)d_in[12];
    const float* v_b   = (const float*)d_in[13];
    const float* r_w   = (const float*)d_in[14];
    const float* r_b   = (const float*)d_in[15];
    const float* r_gate= (const float*)d_in[16];
    float* out = (float*)d_out;

    char* ws = (char*)d_ws;
    size_t off = 0;
    auto alloc = [&](size_t bytes) -> char* {
        char* p = ws + off;
        off = (off + bytes + 255) & ~(size_t)255;
        return p;
    };
    float*  cf    = (float*)alloc(2048 * sizeof(float));
    float*  qsA   = (float*)alloc(NHID * sizeof(float));
    float*  ksA   = (float*)alloc(NHID * sizeof(float));
    float*  vsA   = (float*)alloc(NHID * sizeof(float));
    half_t* wq16  = (half_t*)alloc((size_t)NHID * NHID * 2);
    half_t* wk16  = (half_t*)alloc((size_t)NHID * NHID * 2);
    half_t* wv16  = (half_t*)alloc((size_t)NHID * NHID * 2);
    half_t* wr16  = (half_t*)alloc((size_t)NHID * 2 * NHID * 2);
    half_t* k16   = (half_t*)alloc((size_t)MROWS * NHID * 2);
    half_t* v16   = (half_t*)alloc((size_t)MROWS * NHID * 2);
    half_t* rcat  = (half_t*)alloc((size_t)MROWS * 2 * NHID * 2);  // [mix | q]
    float*  mixf  = (float*)alloc((size_t)MROWS * NHID * sizeof(float));

    // 1) weight conversion to f16
    cvt_f32_f16<<<512, 256, 0, stream>>>(q_w, wq16, NHID * NHID);
    cvt_f32_f16<<<512, 256, 0, stream>>>(k_w, wk16, NHID * NHID);
    cvt_f32_f16<<<512, 256, 0, stream>>>(v_w, wv16, NHID * NHID);
    cvt_f32_f16<<<512, 256, 0, stream>>>(r_w, wr16, NHID * 2 * NHID);

    // 2) gates
    gate_cf_kernel<<<2048, 256, 0, stream>>>(vs_p, vq_w, vq_b, cf);
    gate_scales_kernel<<<4, 256, 0, stream>>>(cf, qs_p, ks_p, qsA, ksA, vsA);

    // 3) Q/K/V projections (f32 A staged->f16, WMMA, scaled f16 output)
    dim3 ggrid(NHID / 128, MROWS / 128);  // (8, 64)
    gemm_wmma_kernel<false, false><<<ggrid, 256, 0, stream>>>(
        query, wq16, q_b, qsA, nullptr, rcat + NHID, nullptr, NHID, 2 * NHID);
    gemm_wmma_kernel<false, false><<<ggrid, 256, 0, stream>>>(
        key, wk16, k_b, ksA, nullptr, k16, nullptr, NHID, NHID);
    gemm_wmma_kernel<false, false><<<ggrid, 256, 0, stream>>>(
        value, wv16, v_b, vsA, nullptr, v16, nullptr, NHID, NHID);

    // 4) attention
    attn_kernel<<<dim3(SEQ / 64, BATCH * HEADS), 128, 0, stream>>>(
        rcat + NHID, k16, v16, rcat, mixf);

    // 5) output projection over rcat (K=2048) with gelu + gated residual
    gemm_wmma_kernel<true, true><<<ggrid, 256, 0, stream>>>(
        rcat, wr16, r_b, r_gate, mixf, nullptr, out, 2 * NHID, 0);
}
